// VoxelProposalLayerOrigin_69260642615438
// MI455X (gfx1250) — matrix operand order repaired and checked by reading, not compiled, not run on hardware
//
#include <hip/hip_runtime.h>
#include <hip/hip_bf16.h>

// ---------------- constants from the reference ----------------
#define EMBED  128
#define NTOT   (128*128*16)   // 262144 voxels
#define MMAX   93184          // upper bound on Nq (= N_VOL)
#define NVOL   93184
#define NVAL   10080          // 48*160 + 24*80 + 12*40 value tokens

typedef __attribute__((ext_vector_type(16))) __bf16 v16bf;
typedef __attribute__((ext_vector_type(16))) short  v16s;
typedef __attribute__((ext_vector_type(8)))  short  v8s;
typedef __attribute__((ext_vector_type(8)))  float  v8f;

// float -> bf16 bits, round-to-nearest-even
__device__ __forceinline__ unsigned short f2bf(float f) {
  union { float f; unsigned u; } c; c.f = f;
  unsigned r = c.u + 0x7FFFu + ((c.u >> 16) & 1u);
  return (unsigned short)(r >> 16);
}

// ---------------- FOV index compaction ----------------
__global__ void init_kernel(int* mask, int* pos) {
  int i = blockIdx.x * 256 + threadIdx.x;
  mask[i] = 0; pos[i] = -1;
}

__global__ void mark_kernel(const int* __restrict__ vol, int* __restrict__ mask) {
  int i = blockIdx.x * blockDim.x + threadIdx.x;
  if (i >= NVOL) return;
  int x = vol[i*3+0], y = vol[i*3+1], z = vol[i*3+2];
  if (x >= 0 && x < 128 && y >= 0 && y < 128 && z >= 0 && z < 16)
    mask[(x*128 + y)*16 + z] = 1;
}

__global__ void count_kernel(const int* __restrict__ mask, int* __restrict__ bcnt) {
  __shared__ int s[256];
  int t = threadIdx.x;
  int base = blockIdx.x * 1024 + t * 4;
  int c = mask[base] + mask[base+1] + mask[base+2] + mask[base+3];
  s[t] = c; __syncthreads();
  for (int o = 128; o > 0; o >>= 1) { if (t < o) s[t] += s[t+o]; __syncthreads(); }
  if (t == 0) bcnt[blockIdx.x] = s[0];
}

__global__ void scan_kernel(const int* __restrict__ bcnt, int* __restrict__ boff,
                            int* __restrict__ nq) {
  __shared__ int s[256];
  int t = threadIdx.x;
  s[t] = bcnt[t]; __syncthreads();
  for (int o = 1; o < 256; o <<= 1) {
    int v = (t >= o) ? s[t-o] : 0;
    __syncthreads();
    s[t] += v;
    __syncthreads();
  }
  boff[t] = t ? s[t-1] : 0;
  if (t == 255) *nq = s[255];
}

__global__ void emit_kernel(const int* __restrict__ mask, const int* __restrict__ boff,
                            int* __restrict__ idx, int* __restrict__ pos) {
  __shared__ int s[256];
  int t = threadIdx.x;
  int base = blockIdx.x * 1024 + t * 4;
  int m0 = mask[base], m1 = mask[base+1], m2 = mask[base+2], m3 = mask[base+3];
  s[t] = m0 + m1 + m2 + m3; __syncthreads();
  for (int o = 1; o < 256; o <<= 1) {
    int v = (t >= o) ? s[t-o] : 0;
    __syncthreads();
    s[t] += v;
    __syncthreads();
  }
  int n = boff[blockIdx.x] + (t ? s[t-1] : 0);
  if (m0) { idx[n] = base;   pos[base]   = n; n++; }
  if (m1) { idx[n] = base+1; pos[base+1] = n; n++; }
  if (m2) { idx[n] = base+2; pos[base+2] = n; n++; }
  if (m3) { idx[n] = base+3; pos[base+3] = n; }
}

// ---------------- gathers / staging ----------------
__global__ void gather_kernel(const float* __restrict__ scene, const float* __restrict__ spos,
                              const float* __restrict__ refp, const float* __restrict__ vp,
                              const int* __restrict__ idx, const int* __restrict__ nqp,
                              float* __restrict__ qg, float* __restrict__ qpg,
                              float* __restrict__ refx, float* __restrict__ refy,
                              float* __restrict__ scl) {
  int n = blockIdx.x; if (n >= *nqp) return;
  int t = threadIdx.x;
  int lin = idx[n];
  qg [(size_t)n*EMBED + t] = scene[(size_t)lin*EMBED + t];
  qpg[(size_t)n*EMBED + t] = spos [(size_t)lin*EMBED + t];
  if (t == 0) {
    float rx = refp[(size_t)lin*2 + 0], ry = refp[(size_t)lin*2 + 1];
    float vx = vp[0] * (1.0f/1280.0f), vy = vp[1] * (1.0f/384.0f);
    refx[n] = rx; refy[n] = ry;
    scl[n]  = sqrtf((vx-rx)*(vx-rx) + (vy-ry)*(vy-ry));
  }
}

// value tokens, emitted directly as bf16 GEMM A operand
__global__ void build_value_kernel(const float* __restrict__ f0, const float* __restrict__ f1,
                                   const float* __restrict__ f2, unsigned short* __restrict__ vin) {
  int n = blockIdx.x, t = threadIdx.x;   // token, channel
  float val;
  if (n < 7680)      val = f0[(size_t)t*7680 + n];
  else if (n < 9600) val = f1[(size_t)t*1920 + (n - 7680)];
  else               val = f2[(size_t)t*480  + (n - 9600)];
  vin[(size_t)n*EMBED + t] = f2bf(val);
}

// fused (a + b) -> bf16 row-major
__global__ void addcvt_kernel(const float* __restrict__ a, const float* __restrict__ b,
                              unsigned short* __restrict__ o, const int* __restrict__ nqp) {
  int n = blockIdx.x; if (n >= *nqp) return;
  int t = threadIdx.x;
  o[(size_t)n*EMBED + t] = f2bf(a[(size_t)n*EMBED + t] + b[(size_t)n*EMBED + t]);
}

// ---------------- weight pre-swizzle into WMMA B-fragment layout ----------------
// Bsw[((ntile*(K/32)+kb)*32 + lane)*16 + e] = bf16(W[(kb*32 + (lane>>4)*16 + e)*N + ntile*16 + (lane&15)])
__global__ void swizzle_w_kernel(const float* __restrict__ W, unsigned short* __restrict__ Bsw,
                                 int K, int N) {
  int blk = blockIdx.x;              // = ntile*(K/32) + kb
  int KB = K >> 5;
  int ntile = blk / KB, kb = blk - ntile * KB;
  int lane = threadIdx.x;
  int lo = lane & 15, hi = lane >> 4;
  unsigned short* dst = Bsw + (((size_t)blk * 32 + lane) * 16);
  const float* src = W + (size_t)(kb*32 + hi*16) * N + ntile*16 + lo;
#pragma unroll
  for (int e = 0; e < 16; ++e) dst[e] = f2bf(src[(size_t)e * N]);
}

// ---------------- WMMA GEMM: C[M,N] = act(A @ W + bias) (+resid) ----------------
// A: bf16 row-major [M][K]; B: pre-swizzled bf16 fragments. One wave = 16x32 output
// (two 16x16 WMMA tiles sharing the A fragment). f32 accumulate.
// Out-of-range rows are CLAMPED (not skipped): row m of D depends only on row m of A,
// and stores are masked, so no exec-mask branching is needed in the hot loop.
template<int K>
__global__ void gemm_wmma_kernel(const unsigned short* __restrict__ Abf,
                                 const unsigned short* __restrict__ Bsw,
                                 const float* __restrict__ bias,
                                 const float* __restrict__ resid,
                                 float* __restrict__ Cf,            // fp32 out (or null)
                                 unsigned short* __restrict__ Cbf,  // bf16 out (or null)
                                 const int* __restrict__ nqp, int Mmax, int N, int doRelu) {
  const int M = nqp ? *nqp : Mmax;
  if ((int)blockIdx.x * 16 >= M) return;
  const int lane = threadIdx.x & 31;
  const int lo = lane & 15, hi = lane >> 4;
  const int row = min(blockIdx.x * 16 + lo, M - 1);   // clamp: garbage rows masked at store
  const int nt0 = blockIdx.y * 2;
  const int KB = K >> 5;

  const unsigned short* arow = Abf + (size_t)row * K + hi * 16;
  const unsigned short* bp0  = Bsw + (((size_t)nt0       * KB) * 32 + lane) * 16;
  const unsigned short* bp1  = Bsw + (((size_t)(nt0 + 1) * KB) * 32 + lane) * 16;

  v8f acc0 = {}, acc1 = {};
#pragma unroll 4
  for (int kb = 0; kb < KB; ++kb) {
    // A frag (16x32 bf16): element e -> K = e + 16*hi : two 16B chunks
    const v8s* ap = (const v8s*)(arow + kb * 32);
    v8s c1 = ap[0];
    v8s c2 = ap[1];
    v16s as = __builtin_shufflevector(c1, c2, 0,1,2,3,4,5,6,7,8,9,10,11,12,13,14,15);
    v16s b0 = *(const v16s*)(bp0 + (size_t)kb * 32 * 16);
    v16s b1 = *(const v16s*)(bp1 + (size_t)kb * 32 * 16);
    acc0 = __builtin_amdgcn_wmma_f32_16x16x32_bf16(
        false, __builtin_bit_cast(v16bf, as), false, __builtin_bit_cast(v16bf, b0),
        (short)0, acc0, false, false);
    acc1 = __builtin_amdgcn_wmma_f32_16x16x32_bf16(
        false, __builtin_bit_cast(v16bf, as), false, __builtin_bit_cast(v16bf, b1),
        (short)0, acc1, false, false);
  }

#pragma unroll
  for (int half = 0; half < 2; ++half) {
    const v8f acc = half ? acc1 : acc0;
    const int col = (nt0 + half) * 16 + lo;
    const float bcol = bias ? bias[col] : 0.0f;
#pragma unroll
    for (int r = 0; r < 8; ++r) {               // C frag: M = r + 8*hi, N = lo
      int orow = blockIdx.x * 16 + hi * 8 + r;
      if (orow < M) {
        float v = acc[r] + bcol;
        if (doRelu) v = fmaxf(v, 0.0f);
        if (resid)  v += resid[(size_t)orow * N + col];
        if (Cf)  Cf [(size_t)orow * N + col] = v;
        if (Cbf) Cbf[(size_t)orow * N + col] = f2bf(v);
      }
    }
  }
}

// ---------------- layer norm over 128 channels (optional bf16 shadow) ----------------
__global__ void ln_kernel(const float* __restrict__ in, const float* __restrict__ g,
                          const float* __restrict__ b, float* __restrict__ outp,
                          unsigned short* __restrict__ obf, const int* __restrict__ nqp) {
  int n = blockIdx.x; if (n >= *nqp) return;
  int t = threadIdx.x;
  __shared__ float red[128];
  float x = in[(size_t)n*EMBED + t];
  red[t] = x; __syncthreads();
  for (int o = 64; o > 0; o >>= 1) { if (t < o) red[t] += red[t+o]; __syncthreads(); }
  float mean = red[0] * (1.0f/128.0f);
  __syncthreads();
  float d = x - mean;
  red[t] = d * d; __syncthreads();
  for (int o = 64; o > 0; o >>= 1) { if (t < o) red[t] += red[t+o]; __syncthreads(); }
  float var = red[0] * (1.0f/128.0f);
  float y = d * rsqrtf(var + 1e-5f) * g[t] + b[t];
  outp[(size_t)n*EMBED + t] = y;
  if (obf) obf[(size_t)n*EMBED + t] = f2bf(y);
}

// ---------------- softmax + bilinear deformable sampling (bf16 output) ----------------
__global__ void sample_kernel(const float* __restrict__ v, const float* __restrict__ offb,
                              const float* __restrict__ awb, const float* __restrict__ refx,
                              const float* __restrict__ refy, const float* __restrict__ scl,
                              unsigned short* __restrict__ attn, const int* __restrict__ nqp) {
  int n = blockIdx.x; if (n >= *nqp) return;
  int t = threadIdx.x;          // 128 threads = channel h*16+d
  __shared__ float sa[96];
  __shared__ float smx[8], ssum[8];
  __shared__ int   sidx[96*4];
  __shared__ float swt[96*4];

  if (t < 96) sa[t] = awb[(size_t)n*96 + t];
  __syncthreads();
  if (t < 8) {                  // per-head softmax stats over 12 (LEVELS*POINTS)
    float mx = -1e30f;
    for (int j = 0; j < 12; ++j) mx = fmaxf(mx, sa[t*12 + j]);
    float s = 0.0f;
    for (int j = 0; j < 12; ++j) s += __expf(sa[t*12 + j] - mx);
    smx[t] = mx; ssum[t] = s;
  }
  __syncthreads();
  if (t < 96) {                 // per (h,l,p): location, corner indices & weights
    int h = t / 12, lp = t % 12, l = lp / 4, p = lp % 4;
    float wgt = __expf(sa[t] - smx[h]) / ssum[h];
    int WL, HL, ST;
    if (l == 0)      { WL = 160; HL = 48; ST = 0;    }
    else if (l == 1) { WL = 80;  HL = 24; ST = 7680; }
    else             { WL = 40;  HL = 12; ST = 9600; }
    int ocol = ((h*3 + l)*4 + p)*2;
    float ox = offb[(size_t)n*192 + ocol];
    float oy = offb[(size_t)n*192 + ocol + 1];
    float rx = refx[n], ry = refy[n], sc = scl[n];
    float lx = rx + ox / (float)WL * sc;
    float ly = ry + oy / (float)HL * sc;
    float x = lx * (float)WL - 0.5f, y = ly * (float)HL - 0.5f;
    float x0 = floorf(x), y0 = floorf(y);
    float wx = x - x0, wy = y - y0;
    int x0i = (int)x0, y0i = (int)y0;
#pragma unroll
    for (int k = 0; k < 4; ++k) {
      int dx = k & 1, dy = k >> 1;
      int xi = x0i + dx, yi = y0i + dy;
      bool inb = (xi >= 0) && (xi < WL) && (yi >= 0) && (yi < HL);
      int xc = min(max(xi, 0), WL-1), yc = min(max(yi, 0), HL-1);
      sidx[t*4 + k] = ST + yc*WL + xc;
      float cw = (dx ? wx : 1.0f - wx) * (dy ? wy : 1.0f - wy);
      swt[t*4 + k] = inb ? cw * wgt : 0.0f;
    }
  }
  __syncthreads();
  int h = t >> 4;
  float acc = 0.0f;
  for (int j = h*12; j < h*12 + 12; ++j) {
#pragma unroll
    for (int k = 0; k < 4; ++k)
      acc += swt[j*4 + k] * v[(size_t)sidx[j*4 + k]*EMBED + t];
  }
  attn[(size_t)n*EMBED + t] = f2bf(acc);
}

// ---------------- final transpose + scatter: out[c][lin] ----------------
__global__ void finalize_kernel(const float* __restrict__ scene, const float* __restrict__ q,
                                const int* __restrict__ pos, float* __restrict__ out) {
  __shared__ float s[32][129];
  __shared__ int sp[32];
  const int t = threadIdx.x;
  const int lin0 = blockIdx.x * 32;
  if (t < 32) sp[t] = pos[lin0 + t];
  __syncthreads();
  for (int r = 0; r < 32; ++r) {
    int p = sp[r];
    const float* src = (p >= 0) ? (q + (size_t)p*EMBED) : (scene + (size_t)(lin0 + r)*EMBED);
    s[r][t] = src[t];
  }
  __syncthreads();
  const int wsub = t >> 5, ln = t & 31;
  for (int cb = 0; cb < 32; ++cb) {
    int c = cb*4 + wsub;
    out[(size_t)c*NTOT + lin0 + ln] = s[ln][c];
  }
}

// ---------------- launcher ----------------
extern "C" void kernel_launch(void* const* d_in, const int* in_sizes, int n_in,
                              void* d_out, int out_size, void* d_ws, size_t ws_size,
                              hipStream_t stream) {
  (void)in_sizes; (void)n_in; (void)out_size; (void)ws_size;
  const float* vp        = (const float*)d_in[0];
  const float* scene_emb = (const float*)d_in[1];
  const float* feat0     = (const float*)d_in[2];
  const float* feat1     = (const float*)d_in[3];
  const float* feat2     = (const float*)d_in[4];
  const float* scene_pos = (const float*)d_in[5];
  const int*   vol_pts   = (const int*)d_in[6];
  const float* ref_pix   = (const float*)d_in[7];
  const float* value_w   = (const float*)d_in[8];
  const float* value_b   = (const float*)d_in[9];
  const float* off_w     = (const float*)d_in[10];
  const float* off_b     = (const float*)d_in[11];
  const float* aw_w      = (const float*)d_in[12];
  const float* aw_b      = (const float*)d_in[13];
  const float* out_w     = (const float*)d_in[14];
  const float* out_b     = (const float*)d_in[15];
  const float* n1_g      = (const float*)d_in[16];
  const float* n1_b      = (const float*)d_in[17];
  const float* n2_g      = (const float*)d_in[18];
  const float* n2_b      = (const float*)d_in[19];
  const float* ffn_w1    = (const float*)d_in[20];
  const float* ffn_b1    = (const float*)d_in[21];
  const float* ffn_w2    = (const float*)d_in[22];
  const float* ffn_b2    = (const float*)d_in[23];
  float* out = (float*)d_out;

  char* wsp = (char*)d_ws;
  auto take = [&](size_t bytes) -> char* {
    char* p = wsp; wsp += (bytes + 255) & ~(size_t)255; return p;
  };
  int*   mask   = (int*)take((size_t)NTOT*4);
  int*   pos    = (int*)take((size_t)NTOT*4);
  int*   bcnt   = (int*)take(256*4);
  int*   boff   = (int*)take(256*4);
  int*   nq     = (int*)take(256);
  int*   idx    = (int*)take((size_t)MMAX*4);
  float* refx   = (float*)take((size_t)MMAX*4);
  float* refy   = (float*)take((size_t)MMAX*4);
  float* scl    = (float*)take((size_t)MMAX*4);
  float* qg     = (float*)take((size_t)MMAX*EMBED*4);
  float* qpg    = (float*)take((size_t)MMAX*EMBED*4);
  float* tmp    = (float*)take((size_t)MMAX*EMBED*4);
  float* vbuf   = (float*)take((size_t)NVAL*EMBED*4);
  unsigned short* vin_bf  = (unsigned short*)take((size_t)NVAL*EMBED*2);
  unsigned short* qin_bf  = (unsigned short*)take((size_t)MMAX*EMBED*2);
  unsigned short* qbf     = (unsigned short*)take((size_t)MMAX*EMBED*2);
  unsigned short* attn_bf = (unsigned short*)take((size_t)MMAX*EMBED*2);
  // swizzled weights (bf16 fragments)
  unsigned short* val_sw  = (unsigned short*)take((size_t)128*128*2);
  unsigned short* off_sw  = (unsigned short*)take((size_t)128*192*2);
  unsigned short* aw_sw   = (unsigned short*)take((size_t)128*96*2);
  unsigned short* out_sw  = (unsigned short*)take((size_t)128*128*2);
  unsigned short* ffn1_sw = (unsigned short*)take((size_t)128*512*2);
  unsigned short* ffn2_sw = (unsigned short*)take((size_t)512*128*2);
  // shared region: {off,aw} (fp32) overlapped with ffn hidden (bf16)
  char* S = take((size_t)MMAX*192*4 + (size_t)MMAX*96*4);
  float*          offb = (float*)S;
  float*          awb  = (float*)(S + (size_t)MMAX*192*4);
  unsigned short* hbuf = (unsigned short*)S;

  // FOV index compaction (ascending, matches jnp.nonzero)
  init_kernel <<<NTOT/256, 256, 0, stream>>>(mask, pos);
  mark_kernel <<<(NVOL+255)/256, 256, 0, stream>>>(vol_pts, mask);
  count_kernel<<<256, 256, 0, stream>>>(mask, bcnt);
  scan_kernel <<<1, 256, 0, stream>>>(bcnt, boff, nq);
  emit_kernel <<<256, 256, 0, stream>>>(mask, boff, idx, pos);

  // weight pre-swizzle (once)
  swizzle_w_kernel<<<(128/16)*(128/32), 32, 0, stream>>>(value_w, val_sw, 128, 128);
  swizzle_w_kernel<<<(192/16)*(128/32), 32, 0, stream>>>(off_w,   off_sw, 128, 192);
  swizzle_w_kernel<<<( 96/16)*(128/32), 32, 0, stream>>>(aw_w,    aw_sw,  128,  96);
  swizzle_w_kernel<<<(128/16)*(128/32), 32, 0, stream>>>(out_w,   out_sw, 128, 128);
  swizzle_w_kernel<<<(512/16)*(128/32), 32, 0, stream>>>(ffn_w1,  ffn1_sw, 128, 512);
  swizzle_w_kernel<<<(128/16)*(512/32), 32, 0, stream>>>(ffn_w2,  ffn2_sw, 512, 128);

  // gathers + value projection (once: identical across layers)
  gather_kernel<<<MMAX, 128, 0, stream>>>(scene_emb, scene_pos, ref_pix, vp, idx, nq,
                                          qg, qpg, refx, refy, scl);
  build_value_kernel<<<NVAL, 128, 0, stream>>>(feat0, feat1, feat2, vin_bf);
  gemm_wmma_kernel<128><<<dim3(NVAL/16, 128/32), 32, 0, stream>>>(
      vin_bf, val_sw, value_b, nullptr, vbuf, nullptr, nullptr, NVAL, 128, 0);

  for (int layer = 0; layer < 3; ++layer) {
    // qin = bf16(q + qpos)
    addcvt_kernel<<<MMAX, 128, 0, stream>>>(qg, qpg, qin_bf, nq);
    // off / aw projections
    gemm_wmma_kernel<128><<<dim3(MMAX/16, 192/32), 32, 0, stream>>>(
        qin_bf, off_sw, off_b, nullptr, offb, nullptr, nq, MMAX, 192, 0);
    gemm_wmma_kernel<128><<<dim3(MMAX/16, 96/32), 32, 0, stream>>>(
        qin_bf, aw_sw, aw_b, nullptr, awb, nullptr, nq, MMAX, 96, 0);
    // softmax + bilinear deformable sampling -> bf16
    sample_kernel<<<MMAX, 128, 0, stream>>>(vbuf, offb, awb, refx, refy, scl, attn_bf, nq);
    // out proj + residual, LN1 (fp32 out + bf16 shadow for FFN1)
    gemm_wmma_kernel<128><<<dim3(MMAX/16, 128/32), 32, 0, stream>>>(
        attn_bf, out_sw, out_b, qg, tmp, nullptr, nq, MMAX, 128, 0);
    ln_kernel<<<MMAX, 128, 0, stream>>>(tmp, n1_g, n1_b, qg, qbf, nq);
    // FFN: hidden stored directly as bf16
    gemm_wmma_kernel<128><<<dim3(MMAX/16, 512/32), 32, 0, stream>>>(
        qbf, ffn1_sw, ffn_b1, nullptr, nullptr, hbuf, nq, MMAX, 512, 1);
    gemm_wmma_kernel<512><<<dim3(MMAX/16, 128/32), 32, 0, stream>>>(
        hbuf, ffn2_sw, ffn_b2, qg, tmp, nullptr, nq, MMAX, 128, 0);
    ln_kernel<<<MMAX, 128, 0, stream>>>(tmp, n2_g, n2_b, qg, nullptr, nq);
  }

  // scatter updated queries into transposed output
  finalize_kernel<<<NTOT/32, 128, 0, stream>>>(scene_emb, qg, pos, out);
}
